// VRWKV_SpatialMix_V7_85066122264772
// MI455X (gfx1250) — matrix operand order repaired and checked by reading, not compiled, not run on hardware
//
#include <hip/hip_runtime.h>
#include <math.h>

// ---------------------------------------------------------------------------
// RWKV-7 spatial mix for MI455X (gfx1250).
// Token-major (B,T,C) activations; A tiles LDS-swizzled in WMMA fragment
// order; B tiles via GLOBAL_LOAD_ASYNC_TO_LDS_B128 with double-buffered LDS
// (one barrier per K-step, async copy overlapped with v_wmma issue).
// ---------------------------------------------------------------------------

namespace {

constexpr int Cdim = 768;
constexpr int Tdim = 2048;
constexpr int Bdim = 2;
constexpr int Hdim = 12;
constexpr int Gdim = 2;
constexpr int CG   = Cdim / Gdim;   // 384
constexpr int DD   = 64;            // decay/aaa/kkk LoRA rank
constexpr int DG   = Cdim / 4;      // 192 gate LoRA rank
constexpr int DL6  = 6 * 64;        // 384 rows of maa_w1
constexpr long CT  = (long)Cdim * Tdim;

typedef __attribute__((ext_vector_type(16))) __bf16 v16bf;
typedef __attribute__((ext_vector_type(8)))  float  v8f;

constexpr int EPI_F32      = 0;  // store f32, token-major (N x M)
constexpr int EPI_TANH_B16 = 1;  // tanh -> bf16, token-major
constexpr int EPI_MIX_B16  = 2;  // x*(1 + mu[row] + acc) -> bf16, token-major
constexpr int EPI_F32_CT   = 3;  // store f32 channel-major (M x T) for d_out

// Branch-free transcendental helpers (lower to v_exp_f32 / v_log_f32).
__device__ inline float tanh_fast(float x) {
  float e = __expf(2.0f * x);
  return 1.0f - 2.0f / (e + 1.0f);          // saturates to +/-1 at large |x|
}
__device__ inline float sigmoid_fast(float z) {
  return 1.0f / (1.0f + __expf(-z));
}
__device__ inline float softplus_fast(float z) {
  return fmaxf(z, 0.0f) + __logf(1.0f + __expf(-fabsf(z)));
}

// ---------------------------------------------------------------------------
// WMMA GEMM: OUT[n][m] = epi( sum_k W[m][k] * X[n][k] ), per batch b.
//   W : fp32 global, row-major ldw (or K-major when WT=true)
//   X : bf16 global, token-major rows of length ldx (row n, col k)
//   OUT: token-major rows of length ldo, except F32_CT (channel-major)
// Block tile 64(M) x 64(N), K-step 32; 8 waves in 4(M) x 2(N) grid.
// Double-buffered LDS: async B copy + A register loads for tile i+1 are
// issued before the WMMAs of tile i; single barrier per K-step.
// ---------------------------------------------------------------------------
template<bool WT, int EPI>
__global__ __launch_bounds__(256)
void wmma_gemm(const float* __restrict__ W, int ldw,
               const __bf16* __restrict__ X, int ldx, long xBatch,
               void* __restrict__ OUT, int ldo, long oBatch,
               int K,
               const float* __restrict__ auxX, long auxBatch,
               const float* __restrict__ auxVec)
{
  const int b  = blockIdx.z;
  const int m0 = blockIdx.y * 64;
  const int n0 = blockIdx.x * 64;
  const __bf16* Xb = X + (long)b * xBatch;

  // A in WMMA-fragment order: As[buf][strip][half*16 + lane16][i].
  __shared__ __align__(16) __bf16 As[2][4][32][16];   // 2 x 4 KB
  // B tile row-major [n][k], row pad 40 keeps 16B alignment per row.
  __shared__ __align__(16) __bf16 Bs[2][64][40];      // 2 x 5 KB

  const int tid  = threadIdx.x;
  const int wave = tid >> 5;
  const int lane = tid & 31;
  const int wm   = wave >> 1;     // 0..3 : 16-row M strip
  const int wn   = wave & 1;      // 0..1 : 32-col N strip
  const int half = lane >> 4;
  const int l16  = lane & 15;

  auto loadA = [&](int k0, float4* wv) {
    #pragma unroll
    for (int cc = 0; cc < 2; ++cc) {
      int cidx = tid * 2 + cc;                  // 512 chunks of 4 elements
      if constexpr (WT) {
        int k = cidx >> 4, m = (cidx & 15) * 4;
        wv[cc] = *(const float4*)&W[(long)(k0 + k) * ldw + (m0 + m)];
      } else {
        int mm = cidx >> 3, k = (cidx & 7) * 4;
        wv[cc] = *(const float4*)&W[(long)(m0 + mm) * ldw + (k0 + k)];
      }
    }
  };
  auto storeA = [&](int buf, const float4* wv) {
    #pragma unroll
    for (int cc = 0; cc < 2; ++cc) {
      int cidx = tid * 2 + cc;
      if constexpr (WT) {
        int k = cidx >> 4, m = (cidx & 15) * 4;
        int hf = k >> 4;
        int ib = k & 15;
        __bf16* dst = &As[buf][m >> 4][hf * 16 + (m & 15)][ib];
        dst[0]  = (__bf16)wv[cc].x; dst[16] = (__bf16)wv[cc].y;
        dst[32] = (__bf16)wv[cc].z; dst[48] = (__bf16)wv[cc].w;
      } else {
        int mm = cidx >> 3, k = (cidx & 7) * 4;
        int hf = k >> 4;
        int ib = k & 15;                        // 4 consecutive i
        __align__(8) __bf16 pk[4] = {(__bf16)wv[cc].x, (__bf16)wv[cc].y,
                                     (__bf16)wv[cc].z, (__bf16)wv[cc].w};
        *(uint2*)&As[buf][mm >> 4][hf * 16 + (mm & 15)][ib] = *(const uint2*)pk;
      }
    }
  };
  auto asyncB = [&](int k0, int buf) {
    int nn = tid >> 2;                          // 0..63
    int kk = (tid & 3) * 8;                     // 0,8,16,24
    unsigned long long ga =
        (unsigned long long)(Xb + (long)(n0 + nn) * ldx + (k0 + kk));
    unsigned lo = (unsigned)(size_t)&Bs[buf][nn][kk];
    asm volatile("global_load_async_to_lds_b128 %0, %1, off"
                 :: "v"(lo), "v"(ga) : "memory");
  };

  v8f acc0 = {};
  v8f acc1 = {};

  // Prologue: tile 0 into buffer 0.
  float4 wv[2];
  asyncB(0, 0);
  loadA(0, wv);
  storeA(0, wv);
  asm volatile("s_wait_asynccnt 0x0" ::: "memory");
  __syncthreads();

  int buf = 0;
  for (int k0 = 0; k0 < K; k0 += 32, buf ^= 1) {
    const bool more = (k0 + 32 < K);
    if (more) {                                 // start tile i+1 (other buf)
      asyncB(k0 + 32, buf ^ 1);
      loadA(k0 + 32, wv);
    }

    v16bf a  = *(const v16bf*)&As[buf][wm][half * 16 + l16][0];
    int nB   = wn * 32 + l16;
    v16bf b0 = *(const v16bf*)&Bs[buf][nB][half * 16];
    v16bf b1 = *(const v16bf*)&Bs[buf][nB + 16][half * 16];

    acc0 = __builtin_amdgcn_wmma_f32_16x16x32_bf16(false, a, false, b0,
                                                   (short)0, acc0, false, false);
    acc1 = __builtin_amdgcn_wmma_f32_16x16x32_bf16(false, a, false, b1,
                                                   (short)0, acc1, false, false);

    if (more) {
      storeA(buf ^ 1, wv);
      asm volatile("s_wait_asynccnt 0x0" ::: "memory");
      __syncthreads();                          // single barrier per K-step
    }
  }

  float*  outF = (float*)OUT  + (long)b * oBatch;
  __bf16* outB = (__bf16*)OUT + (long)b * oBatch;
  const float* axb = (EPI == EPI_MIX_B16) ? auxX + (long)b * auxBatch : nullptr;
  const int rowbase = m0 + wm * 16 + 8 * half;  // acc[r] -> row = rowbase + r

  #pragma unroll
  for (int s = 0; s < 2; ++s) {
    const int col = n0 + wn * 32 + s * 16 + l16;
    v8f acc = s ? acc1 : acc0;
    if constexpr (EPI == EPI_F32) {
      float4 lo = {acc[0], acc[1], acc[2], acc[3]};
      float4 hi = {acc[4], acc[5], acc[6], acc[7]};
      *(float4*)&outF[(long)col * ldo + rowbase]     = lo;
      *(float4*)&outF[(long)col * ldo + rowbase + 4] = hi;
    } else if constexpr (EPI == EPI_F32_CT) {
      #pragma unroll
      for (int r = 0; r < 8; ++r)
        outF[(long)(rowbase + r) * Tdim + col] = acc[r];
    } else if constexpr (EPI == EPI_TANH_B16) {
      __align__(16) __bf16 pk[8];
      #pragma unroll
      for (int r = 0; r < 8; ++r) pk[r] = (__bf16)tanh_fast(acc[r]);
      *(uint4*)&outB[(long)col * ldo + rowbase] = *(const uint4*)pk;
    } else {   // EPI_MIX_B16
      const float* ax = axb + (long)col * Cdim + rowbase;
      float4 x0 = *(const float4*)ax;
      float4 x1 = *(const float4*)(ax + 4);
      float xs[8] = {x0.x, x0.y, x0.z, x0.w, x1.x, x1.y, x1.z, x1.w};
      __align__(16) __bf16 pk[8];
      #pragma unroll
      for (int r = 0; r < 8; ++r)
        pk[r] = (__bf16)(xs[r] * (1.0f + auxVec[rowbase + r] + acc[r]));
      *(uint4*)&outB[(long)col * ldo + rowbase] = *(const uint4*)pk;
    }
  }
}

// ---------------------------------------------------------------------------
// Transpose x (B,C,T) fp32 -> token-major (B,T,C) as fp32 and bf16.
// ---------------------------------------------------------------------------
__global__ __launch_bounds__(256)
void transpose_x_kernel(const float* __restrict__ x,
                        float* __restrict__ xTf, __bf16* __restrict__ xTb)
{
  __shared__ float tile[32][33];
  const int t0 = blockIdx.x * 32, c0 = blockIdx.y * 32, b = blockIdx.z;
  const float* xb = x + (long)b * CT;
  #pragma unroll
  for (int rr = 0; rr < 32; rr += 8) {
    int c = c0 + threadIdx.y + rr;
    tile[threadIdx.y + rr][threadIdx.x] = xb[(long)c * Tdim + t0 + threadIdx.x];
  }
  __syncthreads();
  float*  of = xTf + (long)b * CT;
  __bf16* ob = xTb + (long)b * CT;
  #pragma unroll
  for (int rr = 0; rr < 32; rr += 8) {
    int t = t0 + threadIdx.y + rr;
    float v = tile[threadIdx.x][threadIdx.y + rr];
    of[(long)t * Cdim + c0 + threadIdx.x] = v;
    ob[(long)t * Cdim + c0 + threadIdx.x] = (__bf16)v;
  }
}

// ---------------------------------------------------------------------------
// Pre-WKV elementwise: one block per (b,h,t), 64 threads = head channels.
// Inputs token-major (B,T,C); outputs (B,H,T,64) streams.
// ---------------------------------------------------------------------------
__global__ __launch_bounds__(64)
void prewkv_kernel(const float* __restrict__ R,  const float* __restrict__ Kc,
                   const float* __restrict__ V,  const float* __restrict__ D2,
                   const float* __restrict__ A2, const float* __restrict__ KKl,
                   const float* __restrict__ td, const float* __restrict__ taa,
                   float* __restrict__ rb, float* __restrict__ wb,
                   float* __restrict__ kb, float* __restrict__ vb,
                   float* __restrict__ ab, float* __restrict__ bbv)
{
  const int t = blockIdx.x, h = blockIdx.y, b = blockIdx.z;
  const int i = threadIdx.x;
  const int c = h * 64 + i;
  const long src = ((long)b * Tdim + t) * Cdim + c;

  float r = R[src], k = Kc[src], v = V[src];
  float wlog = -softplus_fast(-(td[c] + D2[src])) - 0.5f;
  float wdec = __expf(-__expf(wlog));
  float kkraw = k + KKl[src];

  __shared__ float red[64];
  red[i] = kkraw * kkraw;
  for (int s = 32; s > 0; s >>= 1) {
    __syncthreads();
    if (i < s) red[i] += red[i + s];
  }
  __syncthreads();
  float kk = kkraw / fmaxf(sqrtf(red[0]), 1e-12f);
  float a  = sigmoid_fast(taa[c] + A2[src]);

  const long dst = (((long)(b * Hdim + h)) * Tdim + t) * 64 + i;
  rb[dst] = r;  wb[dst] = wdec;  kb[dst] = k;
  vb[dst] = v;  ab[dst] = -kk;   bbv[dst] = kk * a;
}

// ---------------------------------------------------------------------------
// WKV-7 scan: one block per (b,h); thread i owns state row S[i][0..63].
// S = S*diag(w) + (S a) b^T + v k^T ; y = S r   -> y token-major (B,T,C)
// ---------------------------------------------------------------------------
__global__ __launch_bounds__(64)
void wkv_kernel(const float* __restrict__ rb, const float* __restrict__ wb,
                const float* __restrict__ kb, const float* __restrict__ vb,
                const float* __restrict__ ab, const float* __restrict__ bbv,
                float* __restrict__ y)
{
  const int bh = blockIdx.x;
  const int b = bh / Hdim, h = bh % Hdim;
  const int i = threadIdx.x;

  float S[64];
  #pragma unroll
  for (int j = 0; j < 64; ++j) S[j] = 0.0f;

  __shared__ float sr[64], sw[64], sk[64], sv[64], sa[64], sb[64];
  const long base = (long)bh * Tdim * 64;

  for (int t = 0; t < Tdim; ++t) {
    const long off = base + (long)t * 64;
    sr[i] = rb[off + i]; sw[i] = wb[off + i]; sk[i] = kb[off + i];
    sv[i] = vb[off + i]; sa[i] = ab[off + i]; sb[i] = bbv[off + i];
    __syncthreads();

    float sad = 0.0f;
    #pragma unroll
    for (int j = 0; j < 64; ++j) sad += S[j] * sa[j];

    const float vi = sv[i];
    float yacc = 0.0f;
    #pragma unroll
    for (int j = 0; j < 64; ++j) {
      float s = S[j] * sw[j] + sad * sb[j] + vi * sk[j];
      S[j] = s;
      yacc += s * sr[j];
    }
    y[((long)b * Tdim + t) * Cdim + h * 64 + i] = yacc;
    __syncthreads();
  }
}

// ---------------------------------------------------------------------------
// Post: RMSNorm(y)*lnx + (sum_i r*k*faaaa)[head]*v, then *g -> bf16 (B,T,C).
// One block per (b,t), 256 threads x 3 channels. All inputs token-major.
// ---------------------------------------------------------------------------
__global__ __launch_bounds__(256)
void post_kernel(const float* __restrict__ y,  const float* __restrict__ Rf,
                 const float* __restrict__ Kf, const float* __restrict__ Vf,
                 const float* __restrict__ Gf, const float* __restrict__ faaaa,
                 const float* __restrict__ lnx, __bf16* __restrict__ yg)
{
  const int t = blockIdx.x, b = blockIdx.y, tid = threadIdx.x;
  __shared__ float p[Cdim];
  __shared__ float red[256];
  __shared__ float rk[Hdim];

  const long rowoff = ((long)b * Tdim + t) * Cdim;
  float ys[3];
  float ss = 0.0f;
  #pragma unroll
  for (int e = 0; e < 3; ++e) {
    int c = tid + 256 * e;
    float yv = y[rowoff + c];
    ys[e] = yv; ss += yv * yv;
    p[c] = Rf[rowoff + c] * Kf[rowoff + c] * faaaa[c];
  }
  red[tid] = ss;
  __syncthreads();
  for (int s = 128; s > 0; s >>= 1) {
    if (tid < s) red[tid] += red[tid + s];
    __syncthreads();
  }
  if (tid < Hdim) {
    float acc = 0.0f;
    for (int j = 0; j < 64; ++j) acc += p[tid * 64 + j];
    rk[tid] = acc;
  }
  __syncthreads();
  float rms = rsqrtf(red[0] / (float)Cdim + 1e-5f);
  #pragma unroll
  for (int e = 0; e < 3; ++e) {
    int c = tid + 256 * e;
    int h = c >> 6;
    float yo = ys[e] * rms * lnx[c] + rk[h] * Vf[rowoff + c];
    yg[rowoff + c] = (__bf16)(yo * Gf[rowoff + c]);
  }
}

} // namespace

// ---------------------------------------------------------------------------
extern "C" void kernel_launch(void* const* d_in, const int* in_sizes, int n_in,
                              void* d_out, int out_size, void* d_ws, size_t ws_size,
                              hipStream_t stream)
{
  (void)in_sizes; (void)n_in; (void)out_size; (void)ws_size;
  const float* x          = (const float*)d_in[0];
  const float* tmv[6]     = { (const float*)d_in[2], (const float*)d_in[3],
                              (const float*)d_in[4], (const float*)d_in[5],
                              (const float*)d_in[6], (const float*)d_in[7] };
  const float* time_decay = (const float*)d_in[8];
  const float* time_faaaa = (const float*)d_in[9];
  const float* time_aaaaa = (const float*)d_in[10];
  const float* maa_w1     = (const float*)d_in[11];
  const float* maa_w2     = (const float*)d_in[12];
  const float* decay_w1   = (const float*)d_in[13];
  const float* decay_w2   = (const float*)d_in[14];
  const float* aaa_w1     = (const float*)d_in[15];
  const float* aaa_w2     = (const float*)d_in[16];
  const float* kkk_w1     = (const float*)d_in[17];
  const float* kkk_w2     = (const float*)d_in[18];
  const float* gate_w1    = (const float*)d_in[19];
  const float* gate_w2    = (const float*)d_in[20];
  const float* w_key      = (const float*)d_in[21];
  const float* w_value    = (const float*)d_in[22];
  const float* w_recept   = (const float*)d_in[23];
  const float* w_output   = (const float*)d_in[24];
  const float* lnx        = (const float*)d_in[25];

  char* wsp = (char*)d_ws;
  auto alloc = [&](size_t bytes) -> char* {
    char* p = wsp;
    wsp += (bytes + 255) & ~(size_t)255;
    return p;
  };
  const long BCT = (long)Bdim * CT;

  __bf16* xTb  = (__bf16*)alloc(BCT * 2);          // x token-major bf16
  float*  xTf  = (float*)alloc(BCT * 4);           // x token-major fp32
  __bf16* TMbf = (__bf16*)alloc((long)Bdim * Tdim * DL6 * 2);
  __bf16* xmix[6];
  for (int n = 0; n < 6; ++n) xmix[n] = (__bf16*)alloc(BCT * 2);
  float* Rf = (float*)alloc(BCT * 4);
  float* Kf = (float*)alloc(BCT * 4);
  float* Vf = (float*)alloc(BCT * 4);
  __bf16* L1bf[3];
  for (int i = 0; i < 3; ++i) L1bf[i] = (__bf16*)alloc((long)Bdim * Tdim * DD * 2);
  __bf16* G1bf = (__bf16*)alloc((long)Bdim * Tdim * DG * 2);
  float* D2  = (float*)alloc(BCT * 4);
  float* A2  = (float*)alloc(BCT * 4);
  float* KKl = (float*)alloc(BCT * 4);
  float* Gf  = (float*)alloc(BCT * 4);
  float* rb  = (float*)alloc(BCT * 4);
  float* wb  = (float*)alloc(BCT * 4);
  float* kb  = (float*)alloc(BCT * 4);
  float* vb  = (float*)alloc(BCT * 4);
  float* ab  = (float*)alloc(BCT * 4);
  float* bbv = (float*)alloc(BCT * 4);
  float* ybuf = (float*)alloc(BCT * 4);
  __bf16* yg  = (__bf16*)alloc(BCT * 2);

  const dim3 blk(256);

  // x -> token-major bf16 + fp32 (xm == x since shift_mode 'none')
  transpose_x_kernel<<<dim3(Tdim / 32, Cdim / 32, Bdim), dim3(32, 8), 0, stream>>>(
      x, xTf, xTb);

  // TM = tanh(maa_w1[384x768] @ x) -> bf16 token-major (B,T,384)
  wmma_gemm<false, EPI_TANH_B16><<<dim3(Tdim/64, DL6/64, Bdim), blk, 0, stream>>>(
      maa_w1, Cdim, xTb, Cdim, CT, TMbf, DL6, (long)Tdim * DL6, Cdim,
      nullptr, 0, nullptr);

  // Six mixes: x{r,w,k,v,a,g} = x*(1 + time_maa_* + maa_w2[n]^T @ TM_n)
  for (int n = 0; n < 6; ++n) {
    wmma_gemm<true, EPI_MIX_B16><<<dim3(Tdim/64, Cdim/64, Bdim), blk, 0, stream>>>(
        maa_w2 + (long)n * 64 * Cdim, Cdim,
        TMbf + n * 64, DL6, (long)Tdim * DL6,
        xmix[n], Cdim, CT, 64, xTf, CT, tmv[n]);
  }
  __bf16 *xr = xmix[0], *xw = xmix[1], *xk = xmix[2];
  __bf16 *xv = xmix[3], *xa = xmix[4], *xg = xmix[5];

  // Grouped 1x1 convs: R, K, V (two 384x384 GEMMs each)
  for (int g = 0; g < Gdim; ++g) {
    int coff  = g * CG;
    long woff = (long)g * CG * CG;
    dim3 gg(Tdim/64, CG/64, Bdim);
    wmma_gemm<false, EPI_F32><<<gg, blk, 0, stream>>>(
        w_recept + woff, CG, xr + coff, Cdim, CT, Rf + coff, Cdim, CT, CG,
        nullptr, 0, nullptr);
    wmma_gemm<false, EPI_F32><<<gg, blk, 0, stream>>>(
        w_key + woff, CG, xk + coff, Cdim, CT, Kf + coff, Cdim, CT, CG,
        nullptr, 0, nullptr);
    wmma_gemm<false, EPI_F32><<<gg, blk, 0, stream>>>(
        w_value + woff, CG, xv + coff, Cdim, CT, Vf + coff, Cdim, CT, CG,
        nullptr, 0, nullptr);
  }

  // Rank-64 LoRAs: decay(xw), aaa(xa), kkk(xk)
  struct Lora { const float* w1; const float* w2; const __bf16* xin; float* out; __bf16* mid; };
  Lora lr[3] = { { decay_w1, decay_w2, xw, D2,  L1bf[0] },
                 { aaa_w1,   aaa_w2,   xa, A2,  L1bf[1] },
                 { kkk_w1,   kkk_w2,   xk, KKl, L1bf[2] } };
  for (int i = 0; i < 3; ++i) {
    wmma_gemm<false, EPI_TANH_B16><<<dim3(Tdim/64, DD/64, Bdim), blk, 0, stream>>>(
        lr[i].w1, Cdim, lr[i].xin, Cdim, CT, lr[i].mid, DD, (long)Tdim * DD, Cdim,
        nullptr, 0, nullptr);
    wmma_gemm<false, EPI_F32><<<dim3(Tdim/64, Cdim/64, Bdim), blk, 0, stream>>>(
        lr[i].w2, DD, lr[i].mid, DD, (long)Tdim * DD, lr[i].out, Cdim, CT, DD,
        nullptr, 0, nullptr);
  }

  // Gate LoRA (rank 192)
  wmma_gemm<false, EPI_TANH_B16><<<dim3(Tdim/64, DG/64, Bdim), blk, 0, stream>>>(
      gate_w1, Cdim, xg, Cdim, CT, G1bf, DG, (long)Tdim * DG, Cdim,
      nullptr, 0, nullptr);
  wmma_gemm<false, EPI_F32><<<dim3(Tdim/64, Cdim/64, Bdim), blk, 0, stream>>>(
      gate_w2, DG, G1bf, DG, (long)Tdim * DG, Gf, Cdim, CT, DG,
      nullptr, 0, nullptr);

  // Elementwise prep for the scan (per-head L2 norm, sigmoid, decay)
  prewkv_kernel<<<dim3(Tdim, Hdim, Bdim), dim3(64), 0, stream>>>(
      Rf, Kf, Vf, D2, A2, KKl, time_decay, time_aaaaa, rb, wb, kb, vb, ab, bbv);

  // Sequential WKV-7 scan (24 independent chains)
  wkv_kernel<<<dim3(Bdim * Hdim), dim3(64), 0, stream>>>(rb, wb, kb, vb, ab, bbv, ybuf);

  // RMSNorm + bonus term + gate -> bf16 token-major
  post_kernel<<<dim3(Tdim, Bdim), dim3(256), 0, stream>>>(
      ybuf, Rf, Kf, Vf, Gf, time_faaaa, lnx, yg);

  // Output grouped conv -> d_out (B,C,T) fp32 (channel-major epilogue)
  for (int g = 0; g < Gdim; ++g) {
    int coff  = g * CG;
    long woff = (long)g * CG * CG;
    wmma_gemm<false, EPI_F32_CT><<<dim3(Tdim/64, CG/64, Bdim), blk, 0, stream>>>(
        w_output + woff, CG, yg + coff, Cdim, CT,
        (float*)d_out + (long)coff * Tdim, Tdim, CT, CG,
        nullptr, 0, nullptr);
  }
}
